// TransMIL1_47021301957026
// MI455X (gfx1250) — matrix operand order repaired and checked by reading, not compiled, not run on hardware
//
#include <hip/hip_runtime.h>
#include <hip/hip_bf16.h>
#include <cmath>

typedef __attribute__((ext_vector_type(16))) _Float16 v16h;
typedef __attribute__((ext_vector_type(8)))  _Float16 v8h;   // 16 B
typedef __attribute__((ext_vector_type(4)))  _Float16 v4h;   // 8 B
typedef __attribute__((ext_vector_type(8)))  float    v8f;
typedef __attribute__((ext_vector_type(4)))  float    v4f;   // 16 B

__device__ __forceinline__ v8f wmma_f16(v16h a, v16h b, v8f c) {
  // D = A(16x32 f16) * B(32x16 f16) + C(16x16 f32)
  return __builtin_amdgcn_wmma_f32_16x16x32_f16(false, a, false, b, (short)0, c,
                                                false, false);
}

// ---------------------------------------------------------------- f32 -> f16
__global__ void cvt_f32_f16(const float* __restrict__ src,
                            _Float16* __restrict__ dst, long count4) {
  long i = (long)blockIdx.x * blockDim.x + threadIdx.x;
  const long stride = (long)gridDim.x * blockDim.x;
  for (; i < count4; i += stride) {
    const v4f v = *(const v4f*)&src[i * 4];
    v4h h;
#pragma unroll
    for (int c = 0; c < 4; ++c) h[c] = (_Float16)v[c];
    *(v4h*)&dst[i * 4] = h;
  }
}

// ---------------------------------------------------------------- WMMA GEMM
// C[M,N] = act(A[M,K] (f32) * B[K,N] (f16) + bias) (+res)
// Block: 128 threads (4 waves). Tile: 64(M) x 64(N), BK=32.
// B tile stored TRANSPOSED in LDS so B-fragment reads are contiguous (b128).
template<int ACT, bool RES>
__global__ __launch_bounds__(128)
void gemm_wmma(const float* __restrict__ A, const _Float16* __restrict__ B,
               const float* __restrict__ bias, const float* __restrict__ res,
               float* __restrict__ C, int M, int N, int K) {
  __shared__ __align__(16) _Float16 sA [64][40];  // [m-row][k]   (80B rows)
  __shared__ __align__(16) _Float16 sBT[64][40];  // [n-col][k]   (80B rows)
  const int tid  = threadIdx.x;
  const int wave = tid >> 5;
  const int lane = tid & 31;
  const int hlf  = lane >> 4;
  const int l16  = lane & 15;
  const int m0   = blockIdx.y * 64;
  const int n0   = blockIdx.x * 64;

  v8f acc[4] = {};

  for (int k0 = 0; k0 < K; k0 += 32) {
    // Stage A tile: 64x32 f32 -> f16. 512 float4 chunks, 4 per thread.
#pragma unroll
    for (int i = 0; i < 4; ++i) {
      const int idx = tid + i * 128;
      const int r = idx >> 3, c4 = (idx & 7) * 4;
      const int gr = m0 + r;
      v4h h = {};
      if (gr < M) {
        const v4f a = *(const v4f*)&A[(long)gr * K + k0 + c4];
#pragma unroll
        for (int c = 0; c < 4; ++c) h[c] = (_Float16)a[c];
      }
      *(v4h*)&sA[r][c4] = h;
    }
    // Stage B tile 32(K)x64(N), transposing into sBT[n][k].
#pragma unroll
    for (int i = 0; i < 2; ++i) {
      const int idx = tid + i * 128;
      const int r = idx >> 3, c8 = (idx & 7) * 8;   // r = k-row, c8 = n-col
      const v8h bv = *(const v8h*)&B[(long)(k0 + r) * N + n0 + c8];
#pragma unroll
      for (int c = 0; c < 8; ++c) sBT[c8 + c][r] = bv[c];
    }
    if (k0 + 32 < K)  // -> global_prefetch_b8
      __builtin_prefetch(&B[(long)(k0 + 32) * N + n0 + (tid & 63)], 0, 1);
    __syncthreads();

    // A fragment (16x32): lanes 0-15 -> K 0-7/16-23, lanes 16-31 -> K 8-15/24-31
    v16h af;
    const int arow = wave * 16 + l16;
    const int akb  = hlf * 8;
#pragma unroll
    for (int j = 0; j < 8; ++j) {
      af[j]     = sA[arow][akb + j];
      af[8 + j] = sA[arow][akb + 16 + j];
    }
    // B fragments: j<8 -> K = hlf*8 + j, j>=8 -> K = 16 + hlf*8 + (j-8) (table order).
    const int bkb = hlf * 8;
#pragma unroll
    for (int t = 0; t < 4; ++t) {
      v16h bf;
      const int bcol = t * 16 + l16;
#pragma unroll
      for (int j = 0; j < 16; ++j) bf[j] = sBT[bcol][bkb + j + (j & 8)];
      acc[t] = wmma_f16(af, bf, acc[t]);
    }
    __syncthreads();
  }

#pragma unroll
  for (int t = 0; t < 4; ++t) {
    const int col = n0 + t * 16 + l16;
#pragma unroll
    for (int r = 0; r < 8; ++r) {
      const int row = m0 + wave * 16 + r + hlf * 8;
      if (row < M) {
        float v = acc[t][r];
        if (bias) v += bias[col];
        if (ACT == 1) v = fmaxf(v, 0.0f);
        else if (ACT == 2) v = 0.5f * v * (1.0f + erff(v * 0.70710678118654752f));
        if (RES) v += res[(long)row * N + col];
        C[(long)row * N + col] = v;
      }
    }
  }
}

// ---------------------------------------------------------------- LayerNorm
__global__ __launch_bounds__(128)
void layernorm_k(const float* __restrict__ X, const float* __restrict__ g,
                 const float* __restrict__ bta, float* __restrict__ Y, int rows) {
  const int lane = threadIdx.x & 31;
  const int row  = blockIdx.x * 4 + (threadIdx.x >> 5);
  if (row >= rows) return;
  const float* x = X + (long)row * 512;
  v4f xv[4];
  float s = 0.f, sq = 0.f;
#pragma unroll
  for (int j = 0; j < 4; ++j) {
    xv[j] = *(const v4f*)&x[lane * 4 + j * 128];
#pragma unroll
    for (int c = 0; c < 4; ++c) { s += xv[j][c]; sq += xv[j][c] * xv[j][c]; }
  }
#pragma unroll
  for (int off = 16; off >= 1; off >>= 1) {
    s  += __shfl_xor(s,  off, 32);
    sq += __shfl_xor(sq, off, 32);
  }
  const float mu   = s * (1.0f / 512.0f);
  const float var  = sq * (1.0f / 512.0f) - mu * mu;
  const float rstd = rsqrtf(var + 1e-5f);
  float* y = Y + (long)row * 512;
#pragma unroll
  for (int j = 0; j < 4; ++j) {
    const int e = lane * 4 + j * 128;
    const v4f gv = *(const v4f*)&g[e];
    const v4f bv = *(const v4f*)&bta[e];
    v4f o;
#pragma unroll
    for (int c = 0; c < 4; ++c) o[c] = (xv[j][c] - mu) * rstd * gv[c] + bv[c];
    *(v4f*)&y[e] = o;
  }
}

// --------------------------------------------------- cls + fc1 output -> x
__global__ __launch_bounds__(128)
void assemble_x(const float* __restrict__ h, const float* __restrict__ cls,
                float* __restrict__ x, int n) {
  const int row = blockIdx.x;            // 0 .. B*n-1
  const int b = row / n, j = row % n;
  const float* src = (j == 0) ? cls : (h + ((long)b * (n - 1) + (j - 1)) * 512);
  const int e = threadIdx.x * 4;         // 128 threads x 4 = 512
  *(v4f*)&x[(long)row * 512 + e] = *(const v4f*)&src[e];
}

// ----------------------------------- Q/K/V prep: mems concat + pos-emb, f16
__global__ __launch_bounds__(128)
void kv_prep(const float* __restrict__ qkv, const float* __restrict__ mem,
             _Float16* __restrict__ Q16, _Float16* __restrict__ K16,
             _Float16* __restrict__ V16, int n, int klen, int mem_len,
             float scale) {
  const int row = blockIdx.x;            // b*klen + j
  const int b = row / klen, j = row % klen;
  const float pos = (float)(klen - 1 - j);
  const int e = threadIdx.x * 4;         // group fully inside sin or cos half
  v4f pe;
#pragma unroll
  for (int c = 0; c < 4; ++c) {
    const int ec = e + c;
    const int ii = (ec < 256) ? ec : (ec - 256);
    const float freq = __expf(-(float)ii * (9.210340371976184f / 256.0f));
    const float ang  = pos * freq;
    pe[c] = (ec < 256) ? __sinf(ang) : __cosf(ang);
  }
  v4f kv, vv;
  if (j < n) {
    const float* qr = qkv + ((long)b * n + j) * 1536;
    kv = *(const v4f*)&qr[512 + e];
    vv = *(const v4f*)&qr[1024 + e];
    const v4f qv = *(const v4f*)&qr[e];
    v4h qh;
#pragma unroll
    for (int c = 0; c < 4; ++c) qh[c] = (_Float16)(qv[c] * scale);
    *(v4h*)&Q16[((long)b * n + j) * 512 + e] = qh;
  } else {
    kv = *(const v4f*)&mem[((long)b * mem_len + (j - n)) * 512 + e];
    vv = kv;
  }
  v4h kh, vh;
#pragma unroll
  for (int c = 0; c < 4; ++c) {
    kh[c] = (_Float16)(kv[c] + pe[c]);
    vh[c] = (_Float16)(vv[c] + pe[c]);
  }
  *(v4h*)&K16[((long)b * klen + j) * 512 + e] = kh;
  *(v4h*)&V16[((long)b * klen + j) * 512 + e] = vh;
}

// -------------------------------------------------- flash attention (wave32)
// Block: 128 threads (4 waves). Each wave owns a 16-query block; all waves
// share LDS-staged K/V tiles (32 keys x 64 dims). Keys streamed in blocks of 32.
// V tile stored TRANSPOSED in LDS so V-fragment reads are contiguous (b128).
__global__ __launch_bounds__(128)
void attn_flash(const _Float16* __restrict__ Q, const _Float16* __restrict__ Kt,
                const _Float16* __restrict__ Vt, float* __restrict__ O,
                int n, int klen) {
  __shared__ __align__(16) _Float16 sQ [64][72];  // 64 q-rows x 64 dims
  __shared__ __align__(16) _Float16 sK [32][72];  // 32 keys  x 64 dims
  __shared__ __align__(16) _Float16 sVT[64][40];  // 64 dims  x 32 keys (transposed)
  __shared__ __align__(16) _Float16 sP [4][16][32]; // per-wave P tile

  const int tid  = threadIdx.x;
  const int wave = tid >> 5;
  const int lane = tid & 31;
  const int hlf  = lane >> 4;
  const int l16  = lane & 15;
  const int bh   = blockIdx.y;
  const int b    = bh >> 3;              // / HEADS
  const int h    = bh & 7;
  const int qblk = blockIdx.x * 64;
  const int q0   = qblk + wave * 16;

  const _Float16* Qb = Q  + ((long)b * n) * 512 + h * 64;
  const _Float16* Kb = Kt + ((long)b * klen) * 512 + h * 64;
  const _Float16* Vb = Vt + ((long)b * klen) * 512 + h * 64;

  // ---- Stage Q tile (64 rows x 64 dims), b128, zero-fill past n.
#pragma unroll
  for (int i = 0; i < 4; ++i) {
    const int idx = tid + i * 128;       // 512 chunks of 8 halves
    const int r = idx >> 3, c8 = (idx & 7) * 8;
    const int qr = qblk + r;
    v8h v = {};
    if (qr < n) v = *(const v8h*)&Qb[(long)qr * 512 + c8];
    *(v8h*)&sQ[r][c8] = v;
  }
  __syncthreads();

  // ---- Q fragments (A layout) for dims 0..31 / 32..63.
  v16h aq0, aq1;
  {
    const int qr = wave * 16 + l16;
    const int kb = hlf * 8;
#pragma unroll
    for (int j = 0; j < 8; ++j) {
      aq0[j]     = sQ[qr][kb + j];
      aq0[8 + j] = sQ[qr][kb + 16 + j];
      aq1[j]     = sQ[qr][32 + kb + j];
      aq1[8 + j] = sQ[qr][32 + kb + 16 + j];
    }
  }

  float m_run[8], l_run[8];
  v8f o[4] = {};
#pragma unroll
  for (int r = 0; r < 8; ++r) { m_run[r] = -1e30f; l_run[r] = 0.0f; }

  for (int kb0 = 0; kb0 < klen; kb0 += 32) {
    __syncthreads();                      // previous-iter LDS reads done
    // ---- Stage K block row-major; V block transposed into sVT[dim][key].
#pragma unroll
    for (int i = 0; i < 2; ++i) {
      const int idx = tid + i * 128;      // 256 chunks of 8 halves
      const int r = idx >> 3, c8 = (idx & 7) * 8;   // r = key, c8 = dim
      const int key = kb0 + r;
      v8h kvv = {}, vvv = {};
      if (key < klen) {
        kvv = *(const v8h*)&Kb[(long)key * 512 + c8];
        vvv = *(const v8h*)&Vb[(long)key * 512 + c8];
      }
      *(v8h*)&sK[r][c8] = kvv;
#pragma unroll
      for (int c = 0; c < 8; ++c) sVT[c8 + c][r] = vvv[c];
    }
    if (kb0 + 32 < klen) {
      __builtin_prefetch(&Kb[(long)(kb0 + 32 + (tid >> 2)) * 512], 0, 1);
      __builtin_prefetch(&Vb[(long)(kb0 + 32 + (tid >> 2)) * 512], 0, 1);
    }
    __syncthreads();

    // ---- S = Q * K^T : two 16x16 accumulators (32 keys).
    v8f S[2];
#pragma unroll
    for (int sub = 0; sub < 2; ++sub) {
      const int krow = sub * 16 + l16;    // key this lane's column maps to
      v16h bk;
#pragma unroll
      for (int j = 0; j < 16; ++j) bk[j] = sK[krow][hlf * 8 + j + (j & 8)];
      v8f s = {};
      s = wmma_f16(aq0, bk, s);
#pragma unroll
      for (int j = 0; j < 16; ++j) bk[j] = sK[krow][32 + hlf * 8 + j + (j & 8)];
      s = wmma_f16(aq1, bk, s);
      if (kb0 + krow >= klen) {           // mask zero-padded keys
#pragma unroll
        for (int r = 0; r < 8; ++r) s[r] = -1e30f;
      }
      S[sub] = s;
    }

    // ---- Online softmax (rows live across 16 lanes of each half-wave).
#pragma unroll
    for (int r = 0; r < 8; ++r) {
      float mx = fmaxf(S[0][r], S[1][r]);
#pragma unroll
      for (int off = 8; off >= 1; off >>= 1) mx = fmaxf(mx, __shfl_xor(mx, off, 16));
      const float mnew = fmaxf(m_run[r], mx);
      const float p0 = __expf(S[0][r] - mnew);
      const float p1 = __expf(S[1][r] - mnew);
      float rs = p0 + p1;
#pragma unroll
      for (int off = 8; off >= 1; off >>= 1) rs += __shfl_xor(rs, off, 16);
      const float sc = __expf(m_run[r] - mnew);
      l_run[r] = l_run[r] * sc + rs;
      m_run[r] = mnew;
#pragma unroll
      for (int t = 0; t < 4; ++t) o[t][r] *= sc;
      const int prow = r + hlf * 8;
      sP[wave][prow][l16]      = (_Float16)p0;
      sP[wave][prow][16 + l16] = (_Float16)p1;
    }
    __syncthreads();                      // P visible across lanes

    // ---- P as A fragment (16 x 32 keys), V as B fragments; O += P*V.
    v16h pa;
    {
      const int kb = hlf * 8;
#pragma unroll
      for (int j = 0; j < 8; ++j) {
        pa[j]     = sP[wave][l16][kb + j];
        pa[8 + j] = sP[wave][l16][kb + 16 + j];
      }
    }
#pragma unroll
    for (int t = 0; t < 4; ++t) {
      v16h vb;
      const int vcol = t * 16 + l16;      // dim
#pragma unroll
      for (int j = 0; j < 16; ++j) vb[j] = sVT[vcol][hlf * 8 + j + (j & 8)];
      o[t] = wmma_f16(pa, vb, o[t]);
    }
  }

#pragma unroll
  for (int t = 0; t < 4; ++t) {
#pragma unroll
    for (int r = 0; r < 8; ++r) {
      const int row = q0 + r + hlf * 8;
      if (row < n)
        O[((long)b * n + row) * 512 + h * 64 + t * 16 + l16] = o[t][r] / l_run[r];
    }
  }
}

// -------------------------------------------- final LN (cls row only) + fc2
__global__ __launch_bounds__(32)
void head_k(const float* __restrict__ x, const float* __restrict__ g,
            const float* __restrict__ bta, const float* __restrict__ w,
            const float* __restrict__ bb, float* __restrict__ out, int n) {
  const int b    = blockIdx.x;
  const int lane = threadIdx.x;
  const float* xr = x + (long)b * n * 512;     // row 0 = cls
  v4f xv[4];
  float s = 0.f, sq = 0.f;
#pragma unroll
  for (int j = 0; j < 4; ++j) {
    xv[j] = *(const v4f*)&xr[lane * 4 + j * 128];
#pragma unroll
    for (int c = 0; c < 4; ++c) { s += xv[j][c]; sq += xv[j][c] * xv[j][c]; }
  }
#pragma unroll
  for (int off = 16; off >= 1; off >>= 1) {
    s  += __shfl_xor(s,  off, 32);
    sq += __shfl_xor(sq, off, 32);
  }
  const float mu = s / 512.0f, var = sq / 512.0f - mu * mu;
  const float rstd = rsqrtf(var + 1e-5f);
  float a0 = 0.f, a1 = 0.f;
#pragma unroll
  for (int j = 0; j < 4; ++j) {
#pragma unroll
    for (int c = 0; c < 4; ++c) {
      const int e = lane * 4 + j * 128 + c;
      const float xn = (xv[j][c] - mu) * rstd * g[e] + bta[e];
      a0 += xn * w[e * 2];
      a1 += xn * w[e * 2 + 1];
    }
  }
#pragma unroll
  for (int off = 16; off >= 1; off >>= 1) {
    a0 += __shfl_xor(a0, off, 32);
    a1 += __shfl_xor(a1, off, 32);
  }
  if (lane == 0) {
    out[b * 2 + 0] = a0 + bb[0];
    out[b * 2 + 1] = a1 + bb[1];
  }
}

// ============================================================== launch =====
extern "C" void kernel_launch(void* const* d_in, const int* in_sizes, int n_in,
                              void* d_out, int out_size, void* d_ws, size_t ws_size,
                              hipStream_t stream) {
  (void)in_sizes; (void)n_in; (void)out_size; (void)ws_size;
  const float* data    = (const float*)d_in[0];
  const float* mems    = (const float*)d_in[1];
  const float* fc1_w   = (const float*)d_in[2];
  const float* fc1_b   = (const float*)d_in[3];
  const float* cls     = (const float*)d_in[4];
  const float* ln1_g   = (const float*)d_in[5];
  const float* ln1_b   = (const float*)d_in[6];
  const float* qkv_w   = (const float*)d_in[7];
  const float* out_w   = (const float*)d_in[8];
  const float* out_b   = (const float*)d_in[9];
  const float* ln2_g   = (const float*)d_in[10];
  const float* ln2_b   = (const float*)d_in[11];
  const float* ff_w1   = (const float*)d_in[12];
  const float* ff_b1   = (const float*)d_in[13];
  const float* ff_w2   = (const float*)d_in[14];
  const float* ff_b2   = (const float*)d_in[15];
  const float* normf_g = (const float*)d_in[16];
  const float* normf_b = (const float*)d_in[17];
  const float* fc2_w   = (const float*)d_in[18];
  const float* fc2_b   = (const float*)d_in[19];
  float* out = (float*)d_out;

  const int B = 2, N = 4096, n = N + 1, mem_len = 8;
  const int klen = n + mem_len;                  // 4105
  const int D = 512, IN = 1024, MLP = 2048;
  const int rows = B * n;                        // 8194

  char* p = (char*)d_ws;
  auto alloc = [&](size_t bytes) -> char* {
    char* r = p;
    p += (bytes + 255) & ~(size_t)255;
    return r;
  };
  float*    x    = (float*)alloc((size_t)rows * D * 4);
  float*    xn   = (float*)alloc((size_t)rows * D * 4);
  float*    hbuf = (float*)alloc((size_t)B * N * D * 4);
  float*    qkv  = (float*)alloc((size_t)rows * 3 * D * 4);
  float*    attn = (float*)alloc((size_t)rows * D * 4);
  float*    ff1  = (float*)alloc((size_t)rows * MLP * 4);
  _Float16* Q16  = (_Float16*)alloc((size_t)rows * D * 2);
  _Float16* K16  = (_Float16*)alloc((size_t)B * klen * D * 2);
  _Float16* V16  = (_Float16*)alloc((size_t)B * klen * D * 2);
  _Float16* wfc1 = (_Float16*)alloc((size_t)IN * D * 2);
  _Float16* wqkv = (_Float16*)alloc((size_t)2 * D * 3 * D * 2);
  _Float16* wout = (_Float16*)alloc((size_t)2 * D * D * 2);
  _Float16* wff1 = (_Float16*)alloc((size_t)2 * D * MLP * 2);
  _Float16* wff2 = (_Float16*)alloc((size_t)2 * MLP * D * 2);

  // Weight conversions (deterministic, every call).
  cvt_f32_f16<<<256, 256, 0, stream>>>(fc1_w, wfc1, (long)IN * D / 4);
  cvt_f32_f16<<<256, 256, 0, stream>>>(qkv_w, wqkv, (long)2 * D * 3 * D / 4);
  cvt_f32_f16<<<256, 256, 0, stream>>>(out_w, wout, (long)2 * D * D / 4);
  cvt_f32_f16<<<256, 256, 0, stream>>>(ff_w1, wff1, (long)2 * D * MLP / 4);
  cvt_f32_f16<<<256, 256, 0, stream>>>(ff_w2, wff2, (long)2 * MLP * D / 4);

  // fc1 + ReLU
  {
    dim3 g(D / 64, (B * N + 63) / 64);
    gemm_wmma<1, false><<<g, 128, 0, stream>>>(data, wfc1, fc1_b, nullptr, hbuf,
                                               B * N, D, IN);
  }
  assemble_x<<<rows, 128, 0, stream>>>(hbuf, cls, x, n);

  const float scale = 0.125f;                    // 64^-0.5
  for (int i = 0; i < 2; ++i) {
    layernorm_k<<<(rows + 3) / 4, 128, 0, stream>>>(x, ln1_g + i * D, ln1_b + i * D,
                                                    xn, rows);
    {
      dim3 g(3 * D / 64, (rows + 63) / 64);
      gemm_wmma<0, false><<<g, 128, 0, stream>>>(xn, wqkv + (size_t)i * D * 3 * D,
                                                 nullptr, nullptr, qkv, rows, 3 * D, D);
    }
    kv_prep<<<B * klen, 128, 0, stream>>>(qkv, mems + (size_t)i * B * mem_len * D,
                                          Q16, K16, V16, n, klen, mem_len, scale);
    {
      dim3 g((n + 63) / 64, B * 8);
      attn_flash<<<g, 128, 0, stream>>>(Q16, K16, V16, attn, n, klen);
    }
    {
      dim3 g(D / 64, (rows + 63) / 64);
      gemm_wmma<0, true><<<g, 128, 0, stream>>>(attn, wout + (size_t)i * D * D,
                                                out_b + i * D, x, x, rows, D, D);
    }
    layernorm_k<<<(rows + 3) / 4, 128, 0, stream>>>(x, ln2_g + i * D, ln2_b + i * D,
                                                    xn, rows);
    {
      dim3 g(MLP / 64, (rows + 63) / 64);
      gemm_wmma<2, false><<<g, 128, 0, stream>>>(xn, wff1 + (size_t)i * D * MLP,
                                                 ff_b1 + i * MLP, nullptr, ff1,
                                                 rows, MLP, D);
    }
    {
      dim3 g(D / 64, (rows + 63) / 64);
      gemm_wmma<0, true><<<g, 128, 0, stream>>>(ff1, wff2 + (size_t)i * MLP * D,
                                                ff_b2 + i * D, x, x, rows, D, MLP);
    }
  }
  head_k<<<B, 32, 0, stream>>>(x, normf_g, normf_b, fc2_w, fc2_b, out, n);
}
